// QuantizedAttention_12171937317141
// MI455X (gfx1250) — 8-bit matrix operand order repaired and checked by reading, compiled, not run on hardware
//
#include <hip/hip_runtime.h>
#include <hip/hip_bf16.h>

typedef __attribute__((ext_vector_type(8))) int   v8i;
typedef __attribute__((ext_vector_type(8))) float v8f;
typedef __attribute__((ext_vector_type(2))) float v2f;

#define BSZ 2
#define NTOK 2048
#define CDIM 1152
#define NH 16
#define HD 72
#define KP 128
#define DP 80
#define C3 3456
#define ROWS 4096

#if __has_builtin(__builtin_amdgcn_wmma_f32_16x16x4_f32)
#define HAVE_WMMA_F32X4 1
#else
#define HAVE_WMMA_F32X4 0
#endif

__device__ __forceinline__ v8i load_a8(const signed char* rowp, int hi, int kbase) {
  union { v8i v; unsigned long long q[4]; } u;
  const char* p = (const char*)rowp + kbase + hi * 8;
  u.q[0] = *(const unsigned long long*)(p);
  u.q[1] = *(const unsigned long long*)(p + 16);
  u.q[2] = *(const unsigned long long*)(p + 32);
  u.q[3] = *(const unsigned long long*)(p + 48);
  return u.v;
}
__device__ __forceinline__ v8i load_b8(const signed char* colp, int hi) {
  union { v8i v; unsigned long long q[4]; } u;
  const char* p = (const char*)colp + hi * 8;
  u.q[0] = *(const unsigned long long*)(p);
  u.q[1] = *(const unsigned long long*)(p + 16);
  u.q[2] = *(const unsigned long long*)(p + 32);
  u.q[3] = *(const unsigned long long*)(p + 48);
  return u.v;
}

__global__ void k_gemm_f32(const float* __restrict__ A, const float* __restrict__ W,
                           const float* __restrict__ bias, float* __restrict__ C,
                           int M, int N, int K) {
#if HAVE_WMMA_F32X4
  const int wave = threadIdx.x >> 5, lane = threadIdx.x & 31;
  const int ln = lane & 15, hi = lane >> 4, koff = hi * 2;
  const int r0 = blockIdx.y * 32;
  const int c0 = blockIdx.x * 128 + wave * 16;
  const float* arow0 = A + (size_t)(r0 + ln) * K;
  const float* arow1 = arow0 + (size_t)16 * K;
  const float* brow  = W + (size_t)(c0 + ln) * K;
  v8f acc0 = {}, acc1 = {};
  for (int k = 0; k < K; k += 4) {
    __builtin_prefetch(brow + k + 256, 0, 1);
    __builtin_prefetch(arow0 + k + 256, 0, 1);
    v2f bv  = *(const v2f*)(brow + k + koff);
    v2f a0v = *(const v2f*)(arow0 + k + koff);
    v2f a1v = *(const v2f*)(arow1 + k + koff);
    acc0 = __builtin_amdgcn_wmma_f32_16x16x4_f32(false, a0v, false, bv, (short)0, acc0, false, false);
    acc1 = __builtin_amdgcn_wmma_f32_16x16x4_f32(false, a1v, false, bv, (short)0, acc1, false, false);
  }
  const int cc = c0 + ln;
  const float bb = bias ? bias[cc] : 0.0f;
#pragma unroll
  for (int i = 0; i < 8; ++i) {
    C[(size_t)(r0 + hi * 8 + i) * N + cc]      = acc0[i] + bb;
    C[(size_t)(r0 + 16 + hi * 8 + i) * N + cc] = acc1[i] + bb;
  }
#else
  const int r0 = blockIdx.y * 32, c0 = blockIdx.x * 128;
  for (int e = threadIdx.x; e < 32 * 128; e += 256) {
    const int r = r0 + (e >> 7), c = c0 + (e & 127);
    float dot = 0.0f;
    for (int k = 0; k < K; ++k) dot += A[(size_t)r * K + k] * W[(size_t)c * K + k];
    C[(size_t)r * N + c] = dot + (bias ? bias[c] : 0.0f);
  }
#endif
}

__global__ void k_quant_qk(const float* __restrict__ qkv, signed char* __restrict__ qi8,
                           signed char* __restrict__ ki8, float* __restrict__ sq,
                           float* __restrict__ sk) {
  const int wave = threadIdx.x >> 5, lane = threadIdx.x & 31;
  const int gid = blockIdx.x * 8 + wave;
  const int sel = gid >> 16;
  const int rem = gid & 65535;
  const int bh = rem >> 11, n = rem & 2047;
  const int b = bh >> 4, h = bh & 15;
  const float* src = qkv + ((size_t)(b * NTOK + n)) * C3 + sel * CDIM + h * HD;
  const float v0 = src[lane], v1 = src[lane + 32];
  const float v2 = (lane < 8) ? src[lane + 64] : 0.0f;
  float am = fmaxf(fmaxf(fabsf(v0), fabsf(v1)), fabsf(v2));
  for (int mask = 16; mask; mask >>= 1) am = fmaxf(am, __shfl_xor(am, mask, 32));
  const float amc = fmaxf(am, 1e-8f);
  const float scale = amc * (1.0f / 127.0f);
  const float inv = 127.0f / amc;
  signed char* dst = (sel ? ki8 : qi8) + ((size_t)(bh * NTOK + n)) * KP;
  auto q8 = [&](float x) {
    int t = (int)rintf(x * inv);
    t = t < -128 ? -128 : (t > 127 ? 127 : t);
    return (signed char)t;
  };
  dst[lane] = q8(v0);
  dst[lane + 32] = q8(v1);
  dst[lane + 64] = (lane < 8) ? q8(v2) : (signed char)0;
  dst[lane + 96] = 0;
  if (lane == 0) (sel ? sk : sq)[bh * NTOK + n] = scale;
}

__global__ void k_quant_v(const float* __restrict__ qkv, signed char* __restrict__ vT,
                          float* __restrict__ sv) {
  const int bh = blockIdx.x / DP, d = blockIdx.x % DP;
  const int b = bh >> 4, h = bh & 15;
  signed char* dst = vT + ((size_t)(bh * DP + d)) * NTOK;
  __shared__ float red[256];
  if (d >= HD) {
    for (int n = threadIdx.x; n < NTOK; n += 256) dst[n] = 0;
    if (threadIdx.x == 0) sv[bh * DP + d] = 0.0f;
    return;
  }
  const float* src = qkv + (size_t)b * NTOK * C3 + 2 * CDIM + h * HD + d;
  float am = 0.0f;
  for (int n = threadIdx.x; n < NTOK; n += 256) am = fmaxf(am, fabsf(src[(size_t)n * C3]));
  red[threadIdx.x] = am;
  __syncthreads();
  for (int s = 128; s; s >>= 1) {
    if (threadIdx.x < s) red[threadIdx.x] = fmaxf(red[threadIdx.x], red[threadIdx.x + s]);
    __syncthreads();
  }
  const float amc = fmaxf(red[0], 1e-8f);
  const float inv = 127.0f / amc;
  for (int n = threadIdx.x; n < NTOK; n += 256) {
    int t = (int)rintf(src[(size_t)n * C3] * inv);
    t = t < -128 ? -128 : (t > 127 ? 127 : t);
    dst[n] = (signed char)t;
  }
  if (threadIdx.x == 0) sv[bh * DP + d] = amc * (1.0f / 127.0f);
}

__global__ void k_softmax_stats(const signed char* __restrict__ qi8, const signed char* __restrict__ ki8,
                                const float* __restrict__ sq, const float* __restrict__ sk,
                                float* __restrict__ mrow, float* __restrict__ zrow) {
  const int bh = blockIdx.x;
  const int wave = threadIdx.x >> 5, lane = threadIdx.x & 31;
  const int ln = lane & 15, hi = lane >> 4;
  const int r0 = blockIdx.y * 128 + wave * 16;
  const signed char* qrow = qi8 + ((size_t)(bh * NTOK + r0 + ln)) * KP;
  const v8i A0 = load_a8(qrow, hi, 0);
  const v8i A1 = load_a8(qrow, hi, 64);
  float sqv[8], m[8], Z[8];
#pragma unroll
  for (int i = 0; i < 8; ++i) {
    sqv[i] = sq[bh * NTOK + r0 + hi * 8 + i];
    m[i] = -1e30f;
    Z[i] = 0.0f;
  }
  const float rs = 0.1178511302f;
  for (int j0 = 0; j0 < NTOK; j0 += 16) {
    const int jc = j0 + ln;
    const signed char* kcol = ki8 + ((size_t)(bh * NTOK + jc)) * KP;
    const v8i B0 = load_b8(kcol, hi);
    const v8i B1 = load_b8(kcol + 64, hi);
    v8i c = {};
    c = __builtin_amdgcn_wmma_i32_16x16x64_iu8(true, A0, true, B0, c, false, false);
    c = __builtin_amdgcn_wmma_i32_16x16x64_iu8(true, A1, true, B1, c, false, false);
    const float skj = sk[bh * NTOK + jc] * rs;
#pragma unroll
    for (int i = 0; i < 8; ++i) {
      const float s = (float)c[i] * sqv[i] * skj;
      const float nm = fmaxf(m[i], s);
      Z[i] = Z[i] * __expf(m[i] - nm) + __expf(s - nm);
      m[i] = nm;
    }
  }
#pragma unroll
  for (int i = 0; i < 8; ++i) {
    float mm = m[i], zz = Z[i];
    for (int mask = 1; mask < 16; mask <<= 1) {
      const float mo = __shfl_xor(mm, mask, 32);
      const float zo = __shfl_xor(zz, mask, 32);
      const float nm = fmaxf(mm, mo);
      zz = zz * __expf(mm - nm) + zo * __expf(mo - nm);
      mm = nm;
    }
    if (ln == 0) {
      const int rr = bh * NTOK + r0 + hi * 8 + i;
      mrow[rr] = mm;
      zrow[rr] = zz;
    }
  }
}

__global__ void k_attn_pv(const signed char* __restrict__ qi8, const signed char* __restrict__ ki8,
                          const signed char* __restrict__ vT, const float* __restrict__ sq,
                          const float* __restrict__ sk, const float* __restrict__ sv,
                          const float* __restrict__ mrow, const float* __restrict__ zrow,
                          float* __restrict__ O) {
  const int bh = blockIdx.x;
  const int wave = threadIdx.x >> 5, lane = threadIdx.x & 31;
  const int ln = lane & 15, hi = lane >> 4;
  const int r0 = blockIdx.y * 128 + wave * 16;
  const int b = bh >> 4, h = bh & 15;
  const signed char* qrow = qi8 + ((size_t)(bh * NTOK + r0 + ln)) * KP;
  const v8i A0 = load_a8(qrow, hi, 0);
  const v8i A1 = load_a8(qrow, hi, 64);
  float sqv[8], mv[8], osc[8];
#pragma unroll
  for (int i = 0; i < 8; ++i) {
    const int rr = bh * NTOK + r0 + hi * 8 + i;
    sqv[i] = sq[rr];
    mv[i] = mrow[rr];
    osc[i] = 1.0f / (255.0f * zrow[rr]);
  }
  const float rs = 0.1178511302f;
  v8i acc[5] = {};
  __shared__ __align__(16) unsigned char lds[8][16 * 64];
  unsigned char* myl = &lds[wave][0];

  for (int j0 = 0; j0 < NTOK; j0 += 64) {
#pragma unroll
    for (int jt = 0; jt < 4; ++jt) {
      const int jc = j0 + jt * 16 + ln;
      const signed char* kcol = ki8 + ((size_t)(bh * NTOK + jc)) * KP;
      const v8i B0 = load_b8(kcol, hi);
      const v8i B1 = load_b8(kcol + 64, hi);
      v8i c = {};
      c = __builtin_amdgcn_wmma_i32_16x16x64_iu8(true, A0, true, B0, c, false, false);
      c = __builtin_amdgcn_wmma_i32_16x16x64_iu8(true, A1, true, B1, c, false, false);
      const float skj = sk[bh * NTOK + jc] * rs;
#pragma unroll
      for (int i = 0; i < 8; ++i) {
        const float s = (float)c[i] * sqv[i] * skj;
        int pi = (int)rintf(255.0f * __expf(s - mv[i]));
        pi = pi < 0 ? 0 : (pi > 255 ? 255 : pi);
        myl[(hi * 8 + i) * 64 + jt * 16 + ln] = (unsigned char)pi;
      }
    }
    __syncthreads();
    v8i Ap;
    {
      union { v8i v; unsigned long long q[4]; } u;
      const unsigned char* p = myl + ln * 64 + hi * 8;
      u.q[0] = *(const unsigned long long*)(p);
      u.q[1] = *(const unsigned long long*)(p + 16);
      u.q[2] = *(const unsigned long long*)(p + 32);
      u.q[3] = *(const unsigned long long*)(p + 48);
      Ap = u.v;
    }
#pragma unroll
    for (int dt = 0; dt < 5; ++dt) {
      const signed char* vcol = vT + ((size_t)(bh * DP + dt * 16 + ln)) * NTOK + j0;
      const v8i Bv = load_b8(vcol, hi);
      acc[dt] = __builtin_amdgcn_wmma_i32_16x16x64_iu8(false, Ap, true, Bv, acc[dt], false, false);
    }
    __syncthreads();
  }
#pragma unroll
  for (int dt = 0; dt < 5; ++dt) {
    const int d = dt * 16 + ln;
    if (d < HD) {
      const float svd = sv[bh * DP + d];
#pragma unroll
      for (int i = 0; i < 8; ++i) {
        const int row = r0 + hi * 8 + i;
        O[((size_t)(b * NTOK + row)) * CDIM + h * HD + d] = (float)acc[dt][i] * osc[i] * svd;
      }
    }
  }
}

extern "C" void kernel_launch(void* const* d_in, const int* in_sizes, int n_in,
                              void* d_out, int out_size, void* d_ws, size_t ws_size,
                              hipStream_t stream) {
  (void)in_sizes; (void)n_in; (void)out_size; (void)ws_size;
  const float* x      = (const float*)d_in[0];
  const float* w_qkv  = (const float*)d_in[1];
  const float* w_proj = (const float*)d_in[2];
  const float* b_proj = (const float*)d_in[3];
  float* out = (float*)d_out;

  char* p = (char*)d_ws;
  auto alloc = [&](size_t bytes) { void* r = (void*)p; p += (bytes + 255) & ~(size_t)255; return r; };
  float*       qkv  = (float*)alloc((size_t)ROWS * C3 * sizeof(float));
  float*       O    = (float*)alloc((size_t)ROWS * CDIM * sizeof(float));
  signed char* qi8  = (signed char*)alloc((size_t)BSZ * NH * NTOK * KP);
  signed char* ki8  = (signed char*)alloc((size_t)BSZ * NH * NTOK * KP);
  signed char* vT   = (signed char*)alloc((size_t)BSZ * NH * DP * NTOK);
  float*       sq   = (float*)alloc((size_t)BSZ * NH * NTOK * sizeof(float));
  float*       sk   = (float*)alloc((size_t)BSZ * NH * NTOK * sizeof(float));
  float*       sv   = (float*)alloc((size_t)BSZ * NH * DP * sizeof(float));
  float*       mrow = (float*)alloc((size_t)BSZ * NH * NTOK * sizeof(float));
  float*       zrow = (float*)alloc((size_t)BSZ * NH * NTOK * sizeof(float));

  k_gemm_f32<<<dim3(C3 / 128, ROWS / 32), 256, 0, stream>>>(x, w_qkv, nullptr, qkv, ROWS, C3, CDIM);
  k_quant_qk<<<(2 * BSZ * NH * NTOK) / 8, 256, 0, stream>>>(qkv, qi8, ki8, sq, sk);
  k_quant_v<<<BSZ * NH * DP, 256, 0, stream>>>(qkv, vT, sv);
  k_softmax_stats<<<dim3(BSZ * NH, NTOK / 128), 256, 0, stream>>>(qi8, ki8, sq, sk, mrow, zrow);
  k_attn_pv<<<dim3(BSZ * NH, NTOK / 128), 256, 0, stream>>>(qi8, ki8, vT, sq, sk, sv, mrow, zrow, O);
  k_gemm_f32<<<dim3(CDIM / 128, ROWS / 32), 256, 0, stream>>>(O, w_proj, b_proj, out, ROWS, CDIM, CDIM);
}
